// VFE_Block_10943576670908
// MI455X (gfx1250) — compile-verified, matrix operand order checked by reading, not run on hardware
//
#include <hip/hip_runtime.h>

typedef __attribute__((ext_vector_type(16))) _Float16 v16h;
typedef __attribute__((ext_vector_type(8)))  float    v8f;
typedef __attribute__((ext_vector_type(4)))  float    v4f;
typedef __attribute__((ext_vector_type(4)))  int      v4i;

#define DGRID 10
#define HGRID 400
#define WGRID 352
#define CFIN  128
#define BATCH 2
#define KVOX  12000
#define TPTS  35
#define NVOX  (BATCH * KVOX)            // 24000
#define NPTS  (NVOX * TPTS)             // 840000
#define DHW   (DGRID * HGRID * WGRID)   // 1,408,000
#define OUT_ELEMS (CFIN * DHW)          // 180,224,000
#define BN_EPS 1e-5f
#define VPB   4                         // voxels per block in layer3

// ---- workspace layout (bytes) ----
#define STATS_OFF 0u                 // 512 floats: sum1[16] sq1[16] scale1[16] shift1[16] @0,16,32,48
                                     //             sum2[64]@64 sq2[64]@128 scale2[64]@192 shift2[64]@256
#define W2T_OFF   4096u              // 2048 f16  : W2^T  [64][32]
#define WFT_OFF   8192u              // 16384 f16 : Wf^T  [128][128]
#define MASK_OFF  40960u             // NPTS f32
#define WIN_OFF   (MASK_OFF + (size_t)NPTS * 4)        // DHW int
#define H1_OFF    (WIN_OFF + (size_t)DHW * 4)          // NPTS*16 f32
#define H2_OFF    (H1_OFF + (size_t)NPTS * 16 * 4)     // NPTS*64 f16

// ---------------------------------------------------------------------------
// CDNA5 async LDS staging (guarded; falls back to synchronous vec copies)
// ---------------------------------------------------------------------------
#if defined(__has_builtin)
#if __has_builtin(__builtin_amdgcn_global_load_async_to_lds_b128)
#define HAVE_ASYNC_COPY 1
#endif
#endif
#ifndef HAVE_ASYNC_COPY
#define HAVE_ASYNC_COPY 0
#endif

#if HAVE_ASYNC_COPY
typedef __attribute__((address_space(1))) v4i g_v4i;   // global 16B vector
typedef __attribute__((address_space(3))) v4i l_v4i;   // LDS 16B vector
#endif

// 128-thread cooperative copy of nbytes (multiple of 2048) global -> LDS
__device__ inline void copy_to_lds_b128(void* dst, const void* src, int nbytes) {
  const int tid = threadIdx.x;
#if HAVE_ASYNC_COPY
  for (int off = tid * 16; off < nbytes; off += 128 * 16) {
    __builtin_amdgcn_global_load_async_to_lds_b128(
        (g_v4i*)((const char*)src + off), (l_v4i*)((char*)dst + off), 0, 0);
  }
#else
  for (int off = tid * 16; off < nbytes; off += 128 * 16) {
    *(v4f*)((char*)dst + off) = *(const v4f*)((const char*)src + off);
  }
#endif
}

__device__ inline void wait_async_copies() {
#if HAVE_ASYNC_COPY
#if __has_builtin(__builtin_amdgcn_s_wait_asynccnt)
  __builtin_amdgcn_s_wait_asynccnt(0);
#else
  asm volatile("s_wait_asynccnt 0x0" ::: "memory");
#endif
#endif
}

// ---------------------------------------------------------------------------
// WMMA helpers
// ---------------------------------------------------------------------------
__device__ inline v8f wmma_f16(v16h a, v16h b, v8f c) {
  return __builtin_amdgcn_wmma_f32_16x16x32_f16(false, a, false, b, (short)0, c,
                                                false, false);
}

// Gather one 16x32 fp16 A/B fragment from row-major LDS tile (ld = row stride).
// Lane layout (CDNA5 16-bit A 16x32): m = lane&15, half = lane>>4,
// vgpr j<4 -> K = half*8 + 2j, vgpr j>=4 -> K = 16 + half*8 + 2(j-4).
__device__ inline v16h load_frag16(const _Float16* base, int ld, int mbase,
                                   int kbase) {
  const int lane = threadIdx.x & 31;
  const _Float16* row =
      base + (mbase + (lane & 15)) * ld + kbase + ((lane >> 4) << 3);
  v16h f;
#pragma unroll
  for (int j = 0; j < 4; ++j) {
    f[2 * j]     = row[2 * j];
    f[2 * j + 1] = row[2 * j + 1];
  }
#pragma unroll
  for (int j = 0; j < 4; ++j) {
    f[8 + 2 * j]     = row[16 + 2 * j];
    f[8 + 2 * j + 1] = row[16 + 2 * j + 1];
  }
  return f;
}

// ---------------------------------------------------------------------------
// 0) zero output grid — non-temporal vec4 stores (don't thrash 192MB L2)
// ---------------------------------------------------------------------------
__global__ void k_zero_out(float* __restrict__ out) {
  const int n4 = OUT_ELEMS / 4;
  const v4f z = {0.f, 0.f, 0.f, 0.f};
  for (int i = blockIdx.x * blockDim.x + threadIdx.x; i < n4;
       i += gridDim.x * blockDim.x) {
    __builtin_nontemporal_store(z, ((v4f*)out) + i);
  }
}

// 1) init winner grid (-1) + zero stats block
__global__ void k_init(int* __restrict__ winner, float* __restrict__ stats) {
  const int i = blockIdx.x * blockDim.x + threadIdx.x;
  if (i < DHW) winner[i] = -1;
  if (i < 512) stats[i] = 0.f;
}

// 2) convert W2 -> W2^T f16, Wf -> Wf^T f16
__global__ void k_prep(const float* __restrict__ W2, const float* __restrict__ Wf,
                       _Float16* __restrict__ w2t, _Float16* __restrict__ wft) {
  const int i = blockIdx.x * blockDim.x + threadIdx.x;
  if (i < 2048) {                       // w2t[o][i32] = W2[i32][o]
    const int o = i >> 5, ii = i & 31;
    w2t[i] = (_Float16)W2[ii * 64 + o];
  } else if (i < 2048 + 16384) {        // wft[d][c] = Wf[c][d]
    const int t = i - 2048;
    const int d = t >> 7, c = t & 127;
    wft[t] = (_Float16)Wf[c * 128 + d];
  }
}

// ---------------------------------------------------------------------------
// 3) layer1: h1 = relu(x*W1 + b1), mask, BN1 partial sums
// ---------------------------------------------------------------------------
__global__ void k_layer1(const float* __restrict__ inp,
                         const float* __restrict__ W1,
                         const float* __restrict__ b1,
                         float* __restrict__ h1, float* __restrict__ maskp,
                         float* __restrict__ stats) {
  __shared__ float s_sum[16], s_sq[16];
  const int tid = threadIdx.x;
  if (tid < 16) { s_sum[tid] = 0.f; s_sq[tid] = 0.f; }
  __syncthreads();

  const int p = blockIdx.x * blockDim.x + tid;
  if (p < NPTS) {
    float x[7];
    float mx = -__builtin_inff();
#pragma unroll
    for (int i = 0; i < 7; ++i) {
      x[i] = inp[(size_t)p * 7 + i];
      mx = fmaxf(mx, x[i]);
    }
    maskp[p] = (mx != 0.f) ? 1.f : 0.f;
#pragma unroll
    for (int o = 0; o < 16; ++o) {
      float h = b1[o];
#pragma unroll
      for (int i = 0; i < 7; ++i) h = fmaf(x[i], W1[i * 16 + o], h);
      h = fmaxf(h, 0.f);
      h1[(size_t)p * 16 + o] = h;
      atomicAdd(&s_sum[o], h);
      atomicAdd(&s_sq[o], h * h);
    }
  }
  __syncthreads();
  if (tid < 16) {
    atomicAdd(&stats[0 + tid], s_sum[tid]);
    atomicAdd(&stats[16 + tid], s_sq[tid]);
  }
}

// 4) finalize BN1 -> scale1/shift1
__global__ void k_fin1(float* __restrict__ stats, const float* __restrict__ g,
                       const float* __restrict__ beta) {
  const int c = threadIdx.x;
  if (c < 16) {
    const float inv = 1.f / (float)NPTS;
    const float mean = stats[0 + c] * inv;
    const float var = stats[16 + c] * inv - mean * mean;
    const float sc = g[c] * rsqrtf(var + BN_EPS);
    stats[32 + c] = sc;
    stats[48 + c] = beta[c] - mean * sc;
  }
}

// 5) claim scatter cells (highest voxel index wins -> deterministic)
__global__ void k_claim(const int* __restrict__ coords, int* __restrict__ winner) {
  const int v = blockIdx.x * blockDim.x + threadIdx.x;
  if (v < NVOX) {
    const int d = coords[v * 3 + 0];
    const int h = coords[v * 3 + 1];
    const int w = coords[v * 3 + 2];
    atomicMax(&winner[d * HGRID * WGRID + h * WGRID + w], v);
  }
}

// ---------------------------------------------------------------------------
// 6) layer2: per voxel — BN1 affine, col-max, concat*mask, WMMA [48x32]x[32x64]
// ---------------------------------------------------------------------------
__global__ __launch_bounds__(128) void k_layer2(
    const float* __restrict__ h1, const float* __restrict__ maskp,
    const _Float16* __restrict__ w2t, const float* __restrict__ b2,
    float* __restrict__ stats, _Float16* __restrict__ h2) {
  __shared__ float s_hn[TPTS][16];
  __shared__ float s_m1[16];
  __shared__ float s_mask[TPTS];
  __shared__ _Float16 As[48][32];
  __shared__ _Float16 Bs[64][32];
  __shared__ float s_sum[64], s_sq[64];

  const int v = blockIdx.x;
  const int tid = threadIdx.x;
  const float* scale1 = stats + 32;
  const float* shift1 = stats + 48;

  copy_to_lds_b128(&Bs[0][0], w2t, 64 * 32 * 2);   // async W2^T stage (4KB)

  if (tid < 64) { s_sum[tid] = 0.f; s_sq[tid] = 0.f; }
  for (int i = tid; i < TPTS * 16; i += 128) {
    const int t = i >> 4, c = i & 15;
    s_hn[t][c] = h1[((size_t)v * TPTS + t) * 16 + c] * scale1[c] + shift1[c];
  }
  if (tid < TPTS) s_mask[tid] = maskp[(size_t)v * TPTS + tid];
  __syncthreads();

  if (tid < 16) {
    float m = -__builtin_inff();
    for (int t = 0; t < TPTS; ++t) m = fmaxf(m, s_hn[t][tid]);
    s_m1[tid] = m;
  }
  __syncthreads();

  for (int i = tid; i < 48 * 32; i += 128) {
    const int t = i >> 5, c = i & 31;
    float val = 0.f;
    if (t < TPTS) val = ((c < 16) ? s_hn[t][c] : s_m1[c - 16]) * s_mask[t];
    As[t][c] = (_Float16)val;
  }
  wait_async_copies();
  __syncthreads();

  const int wave = tid >> 5;
  const int lane = tid & 31;
  const int half = lane >> 4;
  const int nt = wave;                    // 4 waves -> 4 N-tiles
  const int n = nt * 16 + (lane & 15);
  const v16h bfrag = load_frag16(&Bs[0][0], 32, nt * 16, 0);
  const float bias = b2[n];

#pragma unroll
  for (int mt = 0; mt < 3; ++mt) {
    const v16h afrag = load_frag16(&As[0][0], 32, mt * 16, 0);
    v8f acc = {};
    acc = wmma_f16(afrag, bfrag, acc);
    float lsum = 0.f, lsq = 0.f;
#pragma unroll
    for (int r = 0; r < 8; ++r) {
      const int m = mt * 16 + r + 8 * half;
      if (m < TPTS) {
        const float val = fmaxf(acc[r] + bias, 0.f);
        h2[((size_t)v * TPTS + m) * 64 + n] = (_Float16)val;
        lsum += val;
        lsq += val * val;
      }
    }
    atomicAdd(&s_sum[n], lsum);
    atomicAdd(&s_sq[n], lsq);
  }
  __syncthreads();
  if (tid < 64) {
    atomicAdd(&stats[64 + tid], s_sum[tid]);
    atomicAdd(&stats[128 + tid], s_sq[tid]);
  }
}

// 7) finalize BN2 -> scale2/shift2
__global__ void k_fin2(float* __restrict__ stats, const float* __restrict__ g,
                       const float* __restrict__ beta) {
  const int c = threadIdx.x;
  if (c < 64) {
    const float inv = 1.f / (float)NPTS;
    const float mean = stats[64 + c] * inv;
    const float var = stats[128 + c] * inv - mean * mean;
    const float sc = g[c] * rsqrtf(var + BN_EPS);
    stats[192 + c] = sc;
    stats[256 + c] = beta[c] - mean * sc;
  }
}

// ---------------------------------------------------------------------------
// 8) layer3: VPB voxels per block reuse the LDS-resident Wf^T (32KB).
//    BN2 affine, col-max, concat*mask, WMMA [48x128]x[128x128],
//    fused max-over-T, winner-gated scatter into [1,C,D,H,W]
// ---------------------------------------------------------------------------
__global__ __launch_bounds__(128) void k_layer3(
    const _Float16* __restrict__ h2, const float* __restrict__ maskp,
    const _Float16* __restrict__ wft, const float* __restrict__ bf,
    const float* __restrict__ stats, const int* __restrict__ coords,
    const int* __restrict__ winner, float* __restrict__ out) {
  __shared__ float s_hn[TPTS][64];
  __shared__ float s_m2[64];
  __shared__ float s_mask[TPTS];
  __shared__ _Float16 As[48][128];
  __shared__ _Float16 Bs[128][128];
  __shared__ float s_part[3][128];

  const int tid = threadIdx.x;
  const float* scale2 = stats + 192;
  const float* shift2 = stats + 256;
  const int wave = tid >> 5;
  const int lane = tid & 31;
  const int half = lane >> 4;

  // async-stage the shared Wf^T tile; overlap with first voxel's prologue
  copy_to_lds_b128(&Bs[0][0], wft, 128 * 128 * 2);
  bool waited = false;

  for (int vi = 0; vi < VPB; ++vi) {
    const int v = blockIdx.x * VPB + vi;
    const int vnext = (v + 1 < NVOX) ? (v + 1) : v;
    __builtin_prefetch(&h2[((size_t)vnext * TPTS) * 64], 0, 1);

    for (int i = tid; i < TPTS * 64; i += 128) {
      const int t = i >> 6, c = i & 63;
      s_hn[t][c] =
          (float)h2[((size_t)v * TPTS + t) * 64 + c] * scale2[c] + shift2[c];
    }
    if (tid < TPTS) s_mask[tid] = maskp[(size_t)v * TPTS + tid];
    __syncthreads();

    if (tid < 64) {
      float m = -__builtin_inff();
      for (int t = 0; t < TPTS; ++t) m = fmaxf(m, s_hn[t][tid]);
      s_m2[tid] = m;
    }
    __syncthreads();

    for (int i = tid; i < 48 * 128; i += 128) {
      const int t = i >> 7, c = i & 127;
      float val = 0.f;
      if (t < TPTS) val = ((c < 64) ? s_hn[t][c] : s_m2[c - 64]) * s_mask[t];
      As[t][c] = (_Float16)val;
    }
    if (!waited) { wait_async_copies(); waited = true; }
    __syncthreads();

#pragma unroll
    for (int q = 0; q < 2; ++q) {
      const int nt = wave * 2 + q;          // 4 waves x 2 = 8 N-tiles
      const int n = nt * 16 + (lane & 15);
      const float bias = bf[n];
#pragma unroll
      for (int mt = 0; mt < 3; ++mt) {
        v8f acc = {};
#pragma unroll
        for (int kc = 0; kc < 4; ++kc) {
          const v16h a = load_frag16(&As[0][0], 128, mt * 16, kc * 32);
          const v16h b = load_frag16(&Bs[0][0], 128, nt * 16, kc * 32);
          acc = wmma_f16(a, b, acc);
        }
        float lmax = -__builtin_inff();
#pragma unroll
        for (int r = 0; r < 8; ++r) {
          const int m = mt * 16 + r + 8 * half;
          if (m < TPTS) lmax = fmaxf(lmax, acc[r] + bias);
        }
        lmax = fmaxf(lmax, __shfl_xor(lmax, 16, 32));  // merge lane halves
        if (lane < 16) s_part[mt][n] = lmax;
      }
    }
    __syncthreads();

    if (tid < 128) {
      const float vmax =
          fmaxf(s_part[0][tid], fmaxf(s_part[1][tid], s_part[2][tid]));
      const int d = coords[v * 3 + 0];
      const int h = coords[v * 3 + 1];
      const int w = coords[v * 3 + 2];
      const int cell = d * HGRID * WGRID + h * WGRID + w;
      if (winner[cell] == v) out[(size_t)tid * DHW + cell] = vmax;
    }
    __syncthreads();   // s_hn / As reused next voxel
  }
}

// ---------------------------------------------------------------------------
extern "C" void kernel_launch(void* const* d_in, const int* in_sizes, int n_in,
                              void* d_out, int out_size, void* d_ws,
                              size_t ws_size, hipStream_t stream) {
  const float* inp    = (const float*)d_in[0];
  const int*   coords = (const int*)d_in[1];
  const float* W1     = (const float*)d_in[2];
  const float* b1     = (const float*)d_in[3];
  const float* g1     = (const float*)d_in[4];
  const float* beta1  = (const float*)d_in[5];
  const float* W2     = (const float*)d_in[6];
  const float* b2     = (const float*)d_in[7];
  const float* g2     = (const float*)d_in[8];
  const float* beta2  = (const float*)d_in[9];
  const float* Wf     = (const float*)d_in[10];
  const float* bf     = (const float*)d_in[11];
  float* out = (float*)d_out;

  char* ws = (char*)d_ws;
  float*     stats  = (float*)(ws + STATS_OFF);
  _Float16*  w2t    = (_Float16*)(ws + W2T_OFF);
  _Float16*  wft    = (_Float16*)(ws + WFT_OFF);
  float*     maskp  = (float*)(ws + MASK_OFF);
  int*       winner = (int*)(ws + WIN_OFF);
  float*     h1     = (float*)(ws + H1_OFF);
  _Float16*  h2     = (_Float16*)(ws + H2_OFF);

  k_zero_out<<<4096, 256, 0, stream>>>(out);
  k_init<<<(DHW + 255) / 256, 256, 0, stream>>>(winner, stats);
  k_prep<<<(2048 + 16384 + 255) / 256, 256, 0, stream>>>(W2, Wf, w2t, wft);
  k_layer1<<<(NPTS + 255) / 256, 256, 0, stream>>>(inp, W1, b1, h1, maskp,
                                                   stats);
  k_fin1<<<1, 16, 0, stream>>>(stats, g1, beta1);
  k_claim<<<(NVOX + 255) / 256, 256, 0, stream>>>(coords, winner);
  k_layer2<<<NVOX, 128, 0, stream>>>(h1, maskp, w2t, b2, stats, h2);
  k_fin2<<<1, 64, 0, stream>>>(stats, g2, beta2);
  k_layer3<<<NVOX / VPB, 128, 0, stream>>>(h2, maskp, wft, bf, stats, coords,
                                           winner, out);
}
